// GRU_420906795594
// MI455X (gfx1250) — matrix operand order repaired and checked by reading, not compiled, not run on hardware
//
#include <hip/hip_runtime.h>

// ---------------------------------------------------------------------------
// 2-layer custom GRU, persistent bf16-WMMA kernel for gfx1250 (MI455X).
//   BATCH=128, SEQ=256, D_IN=D_H=1024, K=2048.
//   Tiling: 4 row-groups (32 rows = 2 m-tiles) x 32 col-WGs (32 cols = 2
//   n-tiles) = 128 blocks x 256 threads.  B fragment reused across both
//   m-tiles (halves L2 weight traffic vs M=16).
//   Sync: per-group (32 WG) global split barrier (+ s_cluster_barrier hint).
//   Staging: global->LDS async copies (gfx1250), fallback uint4 copies.
// ---------------------------------------------------------------------------

#define SEQ      256
#define DIN      1024
#define DH       1024
#define KDIM     2048
#define NT_TOTAL 64          // 1024/16 N tiles per gate matrix
#define KB_TOTAL 64          // 2048/32 K blocks
#define GROUPS   4
#define ROWS     32          // rows per group (2 WMMA m-tiles)
#define COLWGS   32
#define NCOL     32          // cols per WG (2 WMMA n-tiles)
#define NBLOCKS  (GROUPS * COLWGS)
#define NTHREADS 256
#define CATS     2056        // LDS row stride (2048 + 8 pad) in bf16 elements

#if __has_builtin(__builtin_amdgcn_global_load_async_to_lds_b128)
#define USE_ASYNC_LDS 1
#else
#define USE_ASYNC_LDS 0
#endif

typedef __attribute__((ext_vector_type(8)))  unsigned short v8u;
typedef __attribute__((ext_vector_type(16))) __bf16         v16bf;
typedef __attribute__((ext_vector_type(8)))  float          v8f;

struct U16x2 { v8u lo, hi; };
static_assert(sizeof(U16x2) == 32, "pair must be 32 bytes");

__device__ __forceinline__ v16bf pack16(v8u lo, v8u hi) {
  U16x2 p{lo, hi};
  return __builtin_bit_cast(v16bf, p);
}
__device__ __forceinline__ unsigned short f2bf(float f) {  // RNE f32 -> bf16
  unsigned u = __float_as_uint(f);
  u += 0x7fffu + ((u >> 16) & 1u);
  return (unsigned short)(u >> 16);
}
__device__ __forceinline__ float bf2f(unsigned short h) {
  return __uint_as_float(((unsigned)h) << 16);
}
__device__ __forceinline__ float fast_rcp(float x) {
#if __has_builtin(__builtin_amdgcn_rcpf)
  return __builtin_amdgcn_rcpf(x);   // v_rcp_f32, avoids IEEE div expansion
#else
  return 1.f / x;
#endif
}
__device__ __forceinline__ float sigmoid_(float a) {
  return fast_rcp(1.f + __expf(-a));
}
__device__ __forceinline__ float tanh_(float a) {
  return 1.f - 2.f * fast_rcp(__expf(2.f * a) + 1.f);
}

#if USE_ASYNC_LDS
// Parameter types per hipcc diagnostic: first arg is int4* in the global
// (addrspace(1) / __device__) address space, non-const.
typedef int v4i __attribute__((vector_size(4 * sizeof(int))));
typedef __attribute__((address_space(1))) v4i ga_v4i;
typedef __attribute__((address_space(3))) v4i ls_v4i;
// 16B/lane global->LDS async copy. Integer truncation of a generic LDS
// pointer matches the generic->AS(3) addrspacecast lowering (addr[31:0]).
__device__ __forceinline__ void async_cp_b128(const void* g, void* l) {
  __builtin_amdgcn_global_load_async_to_lds_b128(
      (ga_v4i*)(unsigned long long)g,
      (ls_v4i*)(unsigned)(unsigned long long)l, 0, 0);
}
__device__ __forceinline__ void async_fence() {
#if __has_builtin(__builtin_amdgcn_s_wait_asynccnt)
  __builtin_amdgcn_s_wait_asynccnt(0);
#else
  asm volatile("s_wait_asynccnt 0x0" ::: "memory");
#endif
}
#else
__device__ __forceinline__ void async_fence() {}
#endif

// Split barrier across the 32 workgroups of one row-group (monotonic counter).
__device__ __forceinline__ void group_barrier(unsigned* cnt, unsigned target) {
  __threadfence();
  __syncthreads();
  if (threadIdx.x == 0) {
    __hip_atomic_fetch_add(cnt, 1u, __ATOMIC_RELEASE, __HIP_MEMORY_SCOPE_AGENT);
    while (__hip_atomic_load(cnt, __ATOMIC_ACQUIRE, __HIP_MEMORY_SCOPE_AGENT) < target) {
      __builtin_amdgcn_s_sleep(2);
    }
  }
  __builtin_amdgcn_s_cluster_barrier();  // S_NOP when not dispatched in clusters
  __syncthreads();
  __threadfence();
}

// Dual-m-tile GEMM: B fragment loaded once per k-step, used by two WMMAs.
// KMASK=31 mirrors K in [1024,2048) back onto LDS half0 (layer-1 cat=[h,h]).
// KMASK is a template constant so LDS addressing strength-reduces to
// constant-stride increments in the hot loop.
template <int KMASK>
__device__ __forceinline__ void gemm2(v8f& acc0, v8f& acc1,
                                      const unsigned short* __restrict__ wp,
                                      const unsigned short* ar0,
                                      const unsigned short* ar1,
                                      int kb0, int kb1) {
#pragma unroll 4
  for (int kb = kb0; kb < kb1; ++kb) {
    const int ka = kb & KMASK;
    v8u b0 = *(const v8u*)(wp + (size_t)kb * 512);
    v8u b1 = *(const v8u*)(wp + (size_t)kb * 512 + 8);
    v16bf B = pack16(b0, b1);
    v8u a00 = *(const v8u*)(ar0 + ka * 32);
    v8u a01 = *(const v8u*)(ar0 + ka * 32 + 8);
    acc0 = __builtin_amdgcn_wmma_f32_16x16x32_bf16(false, pack16(a00, a01),
                                                   false, B, (short)0, acc0,
                                                   false, false);
    v8u a10 = *(const v8u*)(ar1 + ka * 32);
    v8u a11 = *(const v8u*)(ar1 + ka * 32 + 8);
    acc1 = __builtin_amdgcn_wmma_f32_16x16x32_bf16(false, pack16(a10, a11),
                                                   false, B, (short)0, acc1,
                                                   false, false);
  }
}

// ---------------------------------------------------------------------------
// Pack W0,W1 (fp32 [3][2048][1024]) into bf16 WMMA B-operand order:
//   record (g6, ntG, kb, lane): 16 elems, K = kb*32 + 16*(lane/16) + e,
//   N = ntG*16 + (lane%16).   6*64*64*32 records * 16 elems = 25.2 MB.
// ---------------------------------------------------------------------------
__global__ __launch_bounds__(256) void gru_pack_w(const float* __restrict__ W0,
                                                  const float* __restrict__ W1,
                                                  unsigned short* __restrict__ Wp) {
  unsigned idx = blockIdx.x * 256u + threadIdx.x;
  const unsigned total = 6u * NT_TOTAL * KB_TOTAL * 32u;
  if (idx >= total) return;
  int lane = idx & 31;
  int kb   = (idx >> 5) & 63;
  int ntG  = (idx >> 11) & 63;
  int g6   = idx >> 17;
  const float* W = (g6 < 3) ? (W0 + (size_t)g6 * KDIM * DH)
                            : (W1 + (size_t)(g6 - 3) * KDIM * DH);
  int n  = ntG * 16 + (lane & 15);
  int k0 = kb * 32 + (lane >> 4) * 16;
  unsigned short* dst = Wp + (size_t)idx * 16;
#pragma unroll
  for (int e = 0; e < 16; ++e) dst[e] = f2bf(W[(size_t)(k0 + e) * DH + n]);
}

__global__ __launch_bounds__(256) void gru_init(float* Hf, unsigned short* Hbf,
                                                unsigned short* HIbf, unsigned* bar) {
  unsigned i = blockIdx.x * 256u + threadIdx.x;
  if (i < 128u * DH) { Hf[i] = 0.f; Hbf[i] = 0; HIbf[i] = 0; }
  if (i < GROUPS * 32u) bar[i] = 0u;
}

// ---------------------------------------------------------------------------
// Persistent recurrent kernel.
// ---------------------------------------------------------------------------
__global__ __launch_bounds__(NTHREADS) void gru_main(
    const float* __restrict__ x, const float* __restrict__ b0,
    const float* __restrict__ b1, const unsigned short* __restrict__ Wp,
    float* __restrict__ Hf, unsigned short* __restrict__ Hbf,
    unsigned short* __restrict__ HIbf, unsigned* __restrict__ bar,
    float* __restrict__ out) {
  __shared__ unsigned short sCat[ROWS * CATS];  // [32][2048+8] bf16 (~128.5 KB)
  __shared__ float sF[ROWS * NCOL];             // forget gate (this WG's cols)
  __shared__ float sZ[3072];                    // K-split partials (12 KB)

  const int tid  = threadIdx.x;
  const int wave = tid >> 5, lane = tid & 31;
  const int hl   = lane >> 4, r15 = lane & 15;
  const int group = blockIdx.x >> 5, wgc = blockIdx.x & 31;
  const int gr0 = group * ROWS, n0 = wgc * NCOL;
  unsigned* gcnt = bar + group * 32;
  unsigned bt = 0;

  const unsigned short* ar0 = sCat + r15 * CATS + hl * 16;  // m-tile 0 A base
  const unsigned short* ar1 = ar0 + 16 * CATS;             // m-tile 1 A base

  // stage ROWS x 1024 bf16 (global, row stride DH) into cat half at dstOff
  auto stage_bf = [&](const unsigned short* __restrict__ src, int dstOff) {
    for (int i = tid; i < ROWS * 128; i += NTHREADS) {
      int rr = i >> 7, c8 = (i & 127) << 3;
      const unsigned short* g = src + (size_t)(gr0 + rr) * DH + c8;
      unsigned short* l = sCat + rr * CATS + dstOff + c8;
#if USE_ASYNC_LDS
      async_cp_b128(g, l);
#else
      *(uint4*)l = *(const uint4*)g;
#endif
    }
  };

#pragma unroll 1
  for (int t = 0; t < SEQ; ++t) {
#pragma unroll 1
    for (int layer = 0; layer < 2; ++layer) {
      const int g3 = layer * 3;
      const float* bv = layer ? b1 : b0;
      const int hOff = layer ? 0 : 1024;  // where h lives for the hi product

      // ---- stage activations ----
      if (layer == 0) {
        const float* xs = x + ((size_t)gr0 * SEQ + t) * DIN;
        for (int i = tid; i < ROWS * 256; i += NTHREADS) {
          int rr = i >> 8, c4 = (i & 255) << 2;
          float4 v = *(const float4*)(xs + (size_t)rr * SEQ * DIN + c4);
          uint2 pk;
          pk.x = (unsigned)f2bf(v.x) | ((unsigned)f2bf(v.y) << 16);
          pk.y = (unsigned)f2bf(v.z) | ((unsigned)f2bf(v.w) << 16);
          *(uint2*)(sCat + rr * CATS + c4) = pk;
        }
        stage_bf(Hbf, 1024);
      } else {
        stage_bf(Hbf, 0);   // h_new(L0) -> half0; GEMM mirrors it for K>=1024
      }
      async_fence();
      __syncthreads();

      // ---- phase A: f = sigmoid(.), i = sigmoid(.), hi = h*i ----
      {
        const int kh = wave >> 2, gate = (wave >> 1) & 1, nt = wave & 1;
        const int ntG = wgc * 2 + nt;
        const unsigned short* wp =
            Wp + ((size_t)((g3 + gate) * NT_TOTAL + ntG) * KB_TOTAL) * 512 + lane * 16;
        v8f acc0 = {}, acc1 = {};
        if (layer == 0)
          gemm2<63>(acc0, acc1, wp, ar0, ar1, kh * 32, kh * 32 + 32);
        else
          gemm2<31>(acc0, acc1, wp, ar0, ar1, kh * 32, kh * 32 + 32);
        const int ca = gate * 2 + nt;
        if (kh == 1) {
          *(v8f*)(sZ + ((ca * 2 + 0) * 32 + lane) * 8) = acc0;
          *(v8f*)(sZ + ((ca * 2 + 1) * 32 + lane) * 8) = acc1;
        }
        __syncthreads();
        if (kh == 0) {
          acc0 += *(const v8f*)(sZ + ((ca * 2 + 0) * 32 + lane) * 8);
          acc1 += *(const v8f*)(sZ + ((ca * 2 + 1) * 32 + lane) * 8);
          const int ncol = nt * 16 + r15, ng = n0 + ncol;
          const float bia = bv[gate * DH + ng];
#pragma unroll
          for (int mt = 0; mt < 2; ++mt) {
            v8f acc = mt ? acc1 : acc0;
#pragma unroll
            for (int v = 0; v < 8; ++v) {
              const int m = mt * 16 + hl * 8 + v;
              float s = sigmoid_(acc[v] + bia);
              if (gate == 0) {
                sF[m * NCOL + ncol] = s;
              } else {
                float h = bf2f(sCat[m * CATS + hOff + ng]);
                HIbf[(size_t)(gr0 + m) * DH + ng] = f2bf(h * s);
              }
            }
          }
        }
      }
      group_barrier(gcnt, (++bt) * COLWGS);  // hi visible group-wide

      // ---- restage cat[:,1024:2048] = h*i ----
      stage_bf(HIbf, 1024);
      async_fence();
      __syncthreads();

      // ---- phase B: z = tanh([x|hi]@W2 + b2); h = (1-f)h + f z ----
      {
        const int kq = wave >> 1, nt = wave & 1;   // 4-way K split
        const int ntG = wgc * 2 + nt;
        const unsigned short* wp =
            Wp + ((size_t)((g3 + 2) * NT_TOTAL + ntG) * KB_TOTAL) * 512 + lane * 16;
        v8f acc0 = {}, acc1 = {};
        gemm2<63>(acc0, acc1, wp, ar0, ar1, kq * 16, kq * 16 + 16);
        if (kq != 0) {
          const int p = kq - 1;
          *(v8f*)(sZ + (((p * 2 + nt) * 2 + 0) * 32 + lane) * 8) = acc0;
          *(v8f*)(sZ + (((p * 2 + nt) * 2 + 1) * 32 + lane) * 8) = acc1;
        }
        __syncthreads();
        if (kq == 0) {
#pragma unroll
          for (int p = 0; p < 3; ++p) {
            acc0 += *(const v8f*)(sZ + (((p * 2 + nt) * 2 + 0) * 32 + lane) * 8);
            acc1 += *(const v8f*)(sZ + (((p * 2 + nt) * 2 + 1) * 32 + lane) * 8);
          }
          const int ncol = nt * 16 + r15, ng = n0 + ncol;
          const float bia = bv[2 * DH + ng];
#pragma unroll
          for (int mt = 0; mt < 2; ++mt) {
            v8f acc = mt ? acc1 : acc0;
#pragma unroll
            for (int v = 0; v < 8; ++v) {
              const int m = mt * 16 + hl * 8 + v;
              float z = tanh_(acc[v] + bia);
              float f = sF[m * NCOL + ncol];
              size_t idx = (size_t)(gr0 + m) * DH + ng;
              float hn = (1.f - f) * Hf[idx] + f * z;
              Hf[idx] = hn;
              Hbf[idx] = f2bf(hn);
            }
          }
        }
      }
      group_barrier(gcnt, (++bt) * COLWGS);  // new h visible group-wide
    }
  }

  // ---- emit final hidden state (fp32 master copy) ----
  for (int i = tid; i < ROWS * NCOL; i += NTHREADS) {
    int rr = i >> 5, cc = i & 31;
    size_t idx = (size_t)(gr0 + rr) * DH + (n0 + cc);
    out[idx] = Hf[idx];
  }
}

// ---------------------------------------------------------------------------
extern "C" void kernel_launch(void* const* d_in, const int* in_sizes, int n_in,
                              void* d_out, int out_size, void* d_ws, size_t ws_size,
                              hipStream_t stream) {
  (void)in_sizes; (void)n_in; (void)out_size; (void)ws_size;
  const float* x  = (const float*)d_in[0];
  const float* W0 = (const float*)d_in[1];
  const float* b0 = (const float*)d_in[2];
  const float* W1 = (const float*)d_in[3];
  const float* b1 = (const float*)d_in[4];

  char* ws = (char*)d_ws;                                   // bytes
  unsigned short* Wp   = (unsigned short*)(ws);             // 25,165,824
  float*          Hf   = (float*)(ws + 25165824);           //    524,288
  unsigned short* Hbf  = (unsigned short*)(ws + 25690112);  //    262,144
  unsigned short* HIbf = (unsigned short*)(ws + 25952256);  //    262,144
  unsigned*       bar  = (unsigned*)(ws + 26214400);        //        512

  gru_init<<<512, 256, 0, stream>>>(Hf, Hbf, HIbf, bar);
  gru_pack_w<<<3072, 256, 0, stream>>>(W0, W1, Wp);
  gru_main<<<NBLOCKS, NTHREADS, 0, stream>>>(x, b0, b1, Wp, Hf, Hbf, HIbf, bar,
                                             (float*)d_out);
}
